// CustomLSTM_68169720922391
// MI455X (gfx1250) — compile-verified, matrix operand order checked by reading, not run on hardware
//
#include <hip/hip_runtime.h>
#include <hip/hip_bf16.h>

typedef __attribute__((ext_vector_type(16))) _Float16 v16h;
typedef __attribute__((ext_vector_type(8)))  _Float16 v8h;
typedef __attribute__((ext_vector_type(8)))  float    v8f;

#define B_   64
#define T_   1024
#define H_   512
#define D0_  256
#define NBLK 32
#define GSTR 68
#define CLIPV 50.0f

// ---------------------------------------------------------------------------
// fragment loader: 16-bit WMMA A/B striping (wave32):
//   lane<16 : row/col = lane,    K = [k0..k0+7] and [k0+16..k0+23]
//   lane>=16: row/col = lane-16, K = [k0+8..k0+15] and [k0+24..k0+31]
// caller resolves the per-lane base pointer; we load two b128 chunks.
// ---------------------------------------------------------------------------
__device__ __forceinline__ v16h ld_frag(const _Float16* p) {
    v8h lo = *(const v8h*)p;
    v8h hi = *(const v8h*)(p + 16);
    return __builtin_shufflevector(lo, hi, 0,1,2,3,4,5,6,7,8,9,10,11,12,13,14,15);
}

__device__ __forceinline__ float sigmoid_f(float x) { return 1.0f / (1.0f + __expf(-x)); }
__device__ __forceinline__ float tanh_f(float x) {
    float e = __expf(-2.0f * x);
    return (1.0f - e) / (1.0f + e);
}
__device__ __forceinline__ float clip_f(float x) { return fminf(CLIPV, fmaxf(-CLIPV, x)); }

// ---------------------------------------------------------------------------
// f32 -> f16 packed converter (8 elems / thread)
// ---------------------------------------------------------------------------
__global__ void cvt_f32_f16(const float* __restrict__ src, _Float16* __restrict__ dst, int n8) {
    int i = blockIdx.x * blockDim.x + threadIdx.x;
    if (i < n8) {
        const float4* s = (const float4*)src + (size_t)i * 2;
        float4 a = s[0], b = s[1];
        v8h o = { (_Float16)a.x, (_Float16)a.y, (_Float16)a.z, (_Float16)a.w,
                  (_Float16)b.x, (_Float16)b.y, (_Float16)b.z, (_Float16)b.w };
        *((v8h*)dst + i) = o;
    }
}

// ---------------------------------------------------------------------------
// Persistent fused LSTM layer.
//  grid = 32 WGs x 256 threads (8 wave32).  WG `wg` owns h-cols [16wg,16wg+16)
//  -> 64 gate columns {512*g + 16wg + j}.  Weight slice resident in LDS.
//  Per step: gates(64x64) = [x_t | h] (64xKT) @ Wslice^T (KTx64) via WMMA,
//  then elementwise LSTM update, h broadcast through global f16 ping-pong,
//  grid barrier via device-scope atomic counter.
// ---------------------------------------------------------------------------
template<int DIN, bool OUTF32>
__global__ __launch_bounds__(256, 1)
void lstm_layer(const _Float16* __restrict__ x_src,   // [B][T][DIN] f16
                const _Float16* __restrict__ w_ih,    // [4H][DIN] f16
                const _Float16* __restrict__ w_hh,    // [4H][H]  f16
                const float*    __restrict__ bias,    // [4H] f32
                _Float16* __restrict__ h_bufs,        // [2][B][H] f16 (zeroed)
                unsigned* __restrict__ counter,       // zeroed
                _Float16* __restrict__ y_f16,         // [B][T][H] (if !OUTF32)
                float*    __restrict__ y_f32,         // [B][T][H] (if OUTF32)
                float*    __restrict__ h_fin,         // [B][H]
                float*    __restrict__ c_fin)         // [B][H]
{
    constexpr int KT   = DIN + H_;
    constexpr int WSTR = KT + 8;          // +8 halfs: 4-dword bank skew per row
    constexpr int NKB  = KT / 32;

    extern __shared__ __align__(16) char smem[];
    _Float16* lds_w = (_Float16*)smem;                                  // [64][WSTR]
    float*    gates = (float*)(smem + (size_t)64 * WSTR * sizeof(_Float16)); // [64][GSTR]

    const int tid  = threadIdx.x;
    const int wg   = blockIdx.x;
    const int wave = tid >> 5;
    const int lane = tid & 31;
    const int mt   = wave & 3;        // M tile (rows 16*mt..16*mt+15 = batch)
    const int nh   = wave >> 2;       // N half (cols 32*nh..32*nh+31)
    const int lr   = lane & 15;
    const int ko   = (lane >> 4) << 3;

    // ---- stage weight slice into LDS (once): row n = gate col 512*(n>>4)+16wg+(n&15)
    for (int idx = tid; idx < 64 * (KT / 8); idx += 256) {
        int n  = idx / (KT / 8);
        int k  = (idx % (KT / 8)) * 8;
        int gc = ((n >> 4) << 9) + (wg << 4) + (n & 15);
        const _Float16* src = (k < DIN) ? (w_ih + (size_t)gc * DIN + k)
                                        : (w_hh + (size_t)gc * H_ + (k - DIN));
        *(v8h*)(lds_w + (size_t)n * WSTR + k) = *(const v8h*)src;
    }
    __syncthreads();

    // ---- per-thread constant biases + cell state registers
    float bi[4], bf[4], bo[4], bg[4], c_reg[4];
#pragma unroll
    for (int r = 0; r < 4; ++r) {
        int e   = tid + 256 * r;
        int col = (wg << 4) + (e & 15);
        bi[r] = bias[col];         bf[r] = bias[512 + col];
        bo[r] = bias[1024 + col];  bg[r] = bias[1536 + col];
        c_reg[r] = 0.0f;
    }

    const int arow = (mt << 4) + lr;

    for (int t = 0; t < T_; ++t) {
        const _Float16* h_prev = h_bufs + (size_t)(t & 1) * (B_ * H_);
        _Float16*       h_next = h_bufs + (size_t)((t + 1) & 1) * (B_ * H_);

        // ---- GEMM: gates(64x64) += A(64xKT) * B(KTx64)
        v8f acc0 = {}; v8f acc1 = {};
        for (int kb = 0; kb < NKB; ++kb) {
            const int k0 = kb * 32;
            const _Float16* ap;
            if (k0 < DIN)
                ap = x_src + (size_t)arow * ((size_t)T_ * DIN) + (size_t)t * DIN + k0 + ko;
            else
                ap = h_prev + (size_t)arow * H_ + (k0 - DIN) + ko;
            v16h a = ld_frag(ap);

            const _Float16* bp = lds_w + (size_t)((nh << 5) + lr) * WSTR + k0 + ko;
            v16h b0v = ld_frag(bp);
            v16h b1v = ld_frag(bp + (size_t)16 * WSTR);

            acc0 = __builtin_amdgcn_wmma_f32_16x16x32_f16(false, a, false, b0v,
                                                          (short)0, acc0, false, false);
            acc1 = __builtin_amdgcn_wmma_f32_16x16x32_f16(false, a, false, b1v,
                                                          (short)0, acc1, false, false);
        }

        // ---- spill accumulators to LDS gates buffer (C/D layout: m=v+8*(lane>>4), n=lane&15)
        {
            const int lm = (lane >> 4) << 3;
            float* gp = gates + (size_t)((mt << 4) + lm) * GSTR + (nh << 5) + lr;
#pragma unroll
            for (int v = 0; v < 8; ++v) {
                gp[(size_t)v * GSTR]      = acc0[v];
                gp[(size_t)v * GSTR + 16] = acc1[v];
            }
        }
        __syncthreads();

        // ---- elementwise LSTM update: 1024 elems (64 rows x 16 h-cols), 4/thread
#pragma unroll
        for (int r = 0; r < 4; ++r) {
            int e   = tid + 256 * r;
            int m   = e >> 4;
            int j   = e & 15;
            int col = (wg << 4) + j;
            const float* gr = gates + (size_t)m * GSTR + j;
            float i_ = sigmoid_f(gr[0]  + bi[r]);
            float f_ = sigmoid_f(gr[16] + bf[r]);
            float o_ = sigmoid_f(gr[32] + bo[r]);
            float g_ = tanh_f   (gr[48] + bg[r]);
            float nc  = f_ * c_reg[r] + i_ * g_;
            float nhh = o_ * tanh_f(nc);          // h uses unclipped c (ref semantics)
            nc  = clip_f(nc);
            nhh = clip_f(nhh);
            c_reg[r] = nc;

            h_next[(size_t)m * H_ + col] = (_Float16)nhh;
            size_t yidx = ((size_t)m * T_ + t) * H_ + col;
            if (OUTF32) y_f32[yidx] = nhh;
            else        y_f16[yidx] = (_Float16)nhh;
            if (t == T_ - 1) {
                h_fin[(size_t)m * H_ + col] = nhh;
                c_fin[(size_t)m * H_ + col] = nc;
            }
        }

        // ---- grid-wide step barrier (release h writes, acquire peers' h)
        if (t != T_ - 1) {
            __threadfence();
            __syncthreads();
            if (tid == 0) {
                __hip_atomic_fetch_add(counter, 1u, __ATOMIC_RELEASE, __HIP_MEMORY_SCOPE_AGENT);
                const unsigned target = (unsigned)(t + 1) * NBLK;
                while (__hip_atomic_load(counter, __ATOMIC_ACQUIRE, __HIP_MEMORY_SCOPE_AGENT) < target)
                    __builtin_amdgcn_s_sleep(8);
            }
            __syncthreads();
        }
    }
}

// ---------------------------------------------------------------------------
// host launcher
// ---------------------------------------------------------------------------
extern "C" void kernel_launch(void* const* d_in, const int* in_sizes, int n_in,
                              void* d_out, int out_size, void* d_ws, size_t ws_size,
                              hipStream_t stream) {
    const float* x     = (const float*)d_in[0];
    const float* w_ih0 = (const float*)d_in[1];
    const float* w_hh0 = (const float*)d_in[2];
    const float* b0    = (const float*)d_in[3];
    const float* w_ih1 = (const float*)d_in[4];
    const float* w_hh1 = (const float*)d_in[5];
    const float* b1    = (const float*)d_in[6];

    char* ws = (char*)d_ws;

    // ---- workspace layout (bytes)
    const size_t SZ_HBUF  = (size_t)2 * B_ * H_ * 2;       // 131072
    const size_t OFF_HB0  = 0;
    const size_t OFF_HB1  = OFF_HB0 + SZ_HBUF;
    const size_t OFF_CNT  = OFF_HB1 + SZ_HBUF;             // counters[2] + pad
    const size_t CTRL_SZ  = OFF_CNT + 256;
    const size_t OFF_XF16 = 524288;
    const size_t SZ_XF16  = (size_t)B_ * T_ * D0_ * 2;     // 32 MB
    const size_t OFF_Y0   = OFF_XF16 + SZ_XF16;
    const size_t SZ_Y0    = (size_t)B_ * T_ * H_ * 2;      // 64 MB
    const size_t OFF_WI0  = OFF_Y0 + SZ_Y0;
    const size_t SZ_WI0   = (size_t)4 * H_ * D0_ * 2;      // 1 MB
    const size_t OFF_WH0  = OFF_WI0 + SZ_WI0;
    const size_t SZ_WHH   = (size_t)4 * H_ * H_ * 2;       // 2 MB
    const size_t OFF_WI1  = OFF_WH0 + SZ_WHH;
    const size_t OFF_WH1  = OFF_WI1 + SZ_WHH;

    _Float16* x_f16  = (_Float16*)(ws + OFF_XF16);
    _Float16* y0_f16 = (_Float16*)(ws + OFF_Y0);
    _Float16* wi0    = (_Float16*)(ws + OFF_WI0);
    _Float16* wh0    = (_Float16*)(ws + OFF_WH0);
    _Float16* wi1    = (_Float16*)(ws + OFF_WI1);
    _Float16* wh1    = (_Float16*)(ws + OFF_WH1);
    _Float16* hbuf0  = (_Float16*)(ws + OFF_HB0);
    _Float16* hbuf1  = (_Float16*)(ws + OFF_HB1);
    unsigned* cnt    = (unsigned*)(ws + OFF_CNT);

    // zero h ping-pong buffers + sync counters (graph-capture safe)
    hipMemsetAsync(ws, 0, CTRL_SZ, stream);

    // ---- f32 -> f16 conversions
    auto cvt = [&](const float* s, _Float16* d, size_t n) {
        int n8 = (int)(n / 8);
        cvt_f32_f16<<<(n8 + 255) / 256, 256, 0, stream>>>(s, d, n8);
    };
    cvt(x,     x_f16, (size_t)B_ * T_ * D0_);
    cvt(w_ih0, wi0,   (size_t)4 * H_ * D0_);
    cvt(w_hh0, wh0,   (size_t)4 * H_ * H_);
    cvt(w_ih1, wi1,   (size_t)4 * H_ * H_);
    cvt(w_hh1, wh1,   (size_t)4 * H_ * H_);

    // ---- output pointers: d_out = [y1 (B*T*H)] [h (2*B*H)] [c (2*B*H)] f32
    float* y1   = (float*)d_out;
    float* hout = y1 + (size_t)B_ * T_ * H_;
    float* cout = hout + (size_t)2 * B_ * H_;

    // ---- dynamic LDS: weights slice + gates buffer
    const size_t GATES_B = (size_t)64 * GSTR * 4;                       // 17408
    const size_t SMEM0 = (size_t)64 * (D0_ + H_ + 8) * 2 + GATES_B;    // 116736
    const size_t SMEM1 = (size_t)64 * (H_  + H_ + 8) * 2 + GATES_B;    // 149504

    lstm_layer<D0_, false><<<NBLK, 256, SMEM0, stream>>>(
        x_f16, wi0, wh0, b0, hbuf0, cnt + 0,
        y0_f16, nullptr, hout + 0, cout + 0);

    lstm_layer<H_, true><<<NBLK, 256, SMEM1, stream>>>(
        y0_f16, wi1, wh1, b1, hbuf1, cnt + 1,
        nullptr, y1, hout + (size_t)B_ * H_, cout + (size_t)B_ * H_);

    (void)in_sizes; (void)n_in; (void)out_size; (void)ws_size;
}
